// MoELayer_67491116089710
// MI455X (gfx1250) — compile-verified, matrix operand order checked by reading, not run on hardware
//
#include <hip/hip_runtime.h>
#include <hip/hip_bf16.h>
#include <math.h>

// ---------------------------------------------------------------------------
// MoE layer for MI455X (gfx1250): bf16 WMMA, token-gathered sparse experts.
// E=8 experts, top-K=2, H=768, I=1152, T=4096 tokens.
// ---------------------------------------------------------------------------

typedef __bf16 bf16;
typedef __attribute__((ext_vector_type(16))) __bf16 v16bf;
typedef __attribute__((ext_vector_type(8)))  __bf16 v8bf;
typedef __attribute__((ext_vector_type(8)))  float  v8f;

constexpr int E_ = 8;
constexpr int H_ = 768;
constexpr int I_ = 1152;
constexpr int T_ = 4096;
constexpr int TB = 16;          // token rows per workgroup (one WMMA M-tile)

// Load a 16x32 bf16 WMMA fragment from a row-major matrix (row stride `ld`
// elements). Lane r (r = lane&15) holds row r; K packing follows the CDNA5
// 16-bit A-matrix layout: elems 0..7 = K(half*8 .. +7), elems 8..15 =
// K(16+half*8 .. +7). Same loader serves the B operand of C = A * W^T
// (W row-major, N x K).
__device__ __forceinline__ v16bf frag_bf16(const bf16* __restrict__ base,
                                           int ld, int lane) {
  const int r = lane & 15, h = lane >> 4;
  const bf16* p = base + (size_t)r * ld + h * 8;
  v8bf lo = *(const v8bf*)(p);
  v8bf hi = *(const v8bf*)(p + 16);
  return __builtin_shufflevector(lo, hi, 0, 1, 2, 3, 4, 5, 6, 7,
                                 8, 9, 10, 11, 12, 13, 14, 15);
}

__device__ __forceinline__ v8f wmma_bf16(v16bf a, v16bf b, v8f c) {
  // D = A * B + C, 16x16x32, f32 accumulate.
  return __builtin_amdgcn_wmma_f32_16x16x32_bf16(
      /*neg_a=*/false, a, /*neg_b=*/false, b,
      /*c_mod=*/(short)0, c, /*reuse_a=*/false, /*reuse_b=*/false);
}

// ---------------------------------------------------------------------------
// Weight pre-processing
// ---------------------------------------------------------------------------

// Per-z-slice 32x32 tiled transpose, fp32 -> bf16 (Wi: (E,H,2I)->(E,2I,H),
// Wo: (E,I,H)->(E,H,I)). Makes every GEMM a contiguous-K  A * W^T.
__global__ __launch_bounds__(256) void transpose_to_bf16(
    const float* __restrict__ src, bf16* __restrict__ dst, int R, int C) {
  __shared__ float tile[32][33];
  const size_t slice = (size_t)R * C;
  src += (size_t)blockIdx.z * slice;
  dst += (size_t)blockIdx.z * slice;
  const int c0 = blockIdx.x * 32, r0 = blockIdx.y * 32;
  const int tx = threadIdx.x & 31, ty = threadIdx.x >> 5;
#pragma unroll
  for (int i = 0; i < 32; i += 8)
    tile[ty + i][tx] = src[(size_t)(r0 + ty + i) * C + c0 + tx];
  __syncthreads();
#pragma unroll
  for (int i = 0; i < 32; i += 8)
    dst[(size_t)(c0 + ty + i) * R + r0 + tx] = (bf16)tile[tx][ty + i];
}

__global__ __launch_bounds__(256) void cvt_bf16(const float* __restrict__ src,
                                                bf16* __restrict__ dst, int n) {
  const int i = blockIdx.x * 256 + threadIdx.x;
  if (i < n) dst[i] = (bf16)src[i];
}

// ---------------------------------------------------------------------------
// Router: sigmoid logits, top-2 (with correction bias), renormalized weights,
// and per-expert gather lists (this is what cuts matrix FLOPs 3x vs dense).
// ---------------------------------------------------------------------------
__global__ __launch_bounds__(256) void routing_kernel(
    const float* __restrict__ x, const float* __restrict__ gw,
    const float* __restrict__ bias, int* __restrict__ cnt,
    int* __restrict__ idx, float* __restrict__ cwb) {
  const int t = blockIdx.x * blockDim.x + threadIdx.x;
  if (t >= T_) return;
  float lg[E_];
#pragma unroll
  for (int e = 0; e < E_; ++e) lg[e] = 0.f;
  const float* xr = x + (size_t)t * H_;
  for (int h = 0; h < H_; ++h) {
    const float xv = xr[h];
#pragma unroll
    for (int e = 0; e < E_; ++e) lg[e] += xv * gw[e * H_ + h];
  }
#pragma unroll
  for (int e = 0; e < E_; ++e) lg[e] = 1.f / (1.f + __expf(-lg[e]));
  int i0 = 0, i1 = 1;
  float b0 = -1e30f, b1 = -1e30f;
#pragma unroll
  for (int e = 0; e < E_; ++e) {
    const float s = lg[e] + bias[e];
    if (s > b0) { b1 = b0; i1 = i0; b0 = s; i0 = e; }
    else if (s > b1) { b1 = s; i1 = e; }
  }
  const float w0 = lg[i0], w1 = lg[i1];
  const float inv = 1.f / (w0 + w1);
  const int p0 = atomicAdd(&cnt[i0], 1);
  idx[i0 * T_ + p0] = t;
  cwb[i0 * T_ + p0] = w0 * inv;
  const int p1 = atomicAdd(&cnt[i1], 1);
  idx[i1 * T_ + p1] = t;
  cwb[i1 * T_ + p1] = w1 * inv;
}

// ---------------------------------------------------------------------------
// Fused MLP: up-proj (2 WMMA accumulators: proj+gate) -> SiLU-gate*cw into an
// LDS activation tile -> down-proj WMMA -> store (shared) / atomic (experts).
// EXPERT_MODE==0: shared expert, identity token map, SiLU on FIRST half.
// EXPERT_MODE==1: routed experts, gathered tokens, SiLU on SECOND half.
// ---------------------------------------------------------------------------
template <int EXPERT_MODE>
__global__ __launch_bounds__(256) void fused_mlp(
    const float* __restrict__ x, const bf16* __restrict__ WiB,
    const bf16* __restrict__ WoB, const int* __restrict__ cnt,
    const int* __restrict__ idx, const float* __restrict__ cwb,
    float* __restrict__ out) {
  __shared__ __align__(16) bf16 s_x[TB * H_];    // 24 KB
  __shared__ __align__(16) bf16 s_act[TB * I_];  // 36 KB
  __shared__ int s_tok[TB];
  __shared__ float s_cw[TB];

  const int t0 = blockIdx.x * TB;
  const bf16* wi = WiB;
  const bf16* wo = WoB;
  int total = T_;
  if (EXPERT_MODE) {
    const int e = blockIdx.y;
    total = cnt[e];
    if (t0 >= total) return;  // uniform early-exit, before any barrier
    wi += (size_t)e * (2 * I_) * H_;
    wo += (size_t)e * H_ * I_;
  }

  const int tid = threadIdx.x;
  if (tid < TB) {
    int tok = 0;
    float w = 0.f;
    const int pos = t0 + tid;
    if (EXPERT_MODE) {
      if (pos < total) {
        tok = idx[blockIdx.y * T_ + pos];
        w = cwb[blockIdx.y * T_ + pos];
      }
    } else {
      tok = pos;
      w = 1.f;
    }
    s_tok[tid] = tok;
    s_cw[tid] = w;
  }
  __syncthreads();

  // Stage the TB gathered x rows into LDS as bf16.
  for (int p = tid; p < TB * (H_ / 4); p += 256) {
    const int row = p / (H_ / 4);
    const int c4 = (p % (H_ / 4)) * 4;
    const float4 v = *(const float4*)(x + (size_t)s_tok[row] * H_ + c4);
    bf16* d = &s_x[row * H_ + c4];
    d[0] = (bf16)v.x; d[1] = (bf16)v.y; d[2] = (bf16)v.z; d[3] = (bf16)v.w;
  }
  __syncthreads();

  const int lane = tid & 31, wave = tid >> 5;

  // ---- UP projection: 72 N-tiles of 16, K over H in steps of 32 ----
  for (int u = wave; u < I_ / 16; u += 8) {
    const int n0 = u * 16;
    v8f accp = {};
    v8f accg = {};
    const bf16* bp = wi + (size_t)n0 * H_;         // proj rows [0, I)
    const bf16* bg = wi + (size_t)(I_ + n0) * H_;  // gate rows [I, 2I)
    for (int kk = 0; kk < H_; kk += 32) {
      v16bf a  = frag_bf16(&s_x[kk], H_, lane);
      v16bf fb = frag_bf16(bp + kk, H_, lane);
      v16bf gb = frag_bf16(bg + kk, H_, lane);
      accp = wmma_bf16(a, fb, accp);
      accg = wmma_bf16(a, gb, accg);
      if (kk + 32 < H_) {
        __builtin_prefetch(bp + (size_t)(lane & 15) * H_ + kk + 32, 0, 1);
        __builtin_prefetch(bg + (size_t)(lane & 15) * H_ + kk + 32, 0, 1);
      }
    }
    const int half = lane >> 4, col = n0 + (lane & 15);
#pragma unroll
    for (int v = 0; v < 8; ++v) {
      const int row = v + 8 * half;
      const float p = accp[v], g = accg[v];
      // shared: silu(first_half)*second ; expert: silu(second_half)*first
      const float act = EXPERT_MODE ? (g / (1.f + __expf(-g))) * p
                                    : (p / (1.f + __expf(-p))) * g;
      s_act[row * I_ + col] = (bf16)(act * s_cw[row]);
    }
  }
  __syncthreads();

  // ---- DOWN projection: 48 H-tiles of 16, K over I in steps of 32 ----
  for (int u = wave; u < H_ / 16; u += 8) {
    const int n0 = u * 16;
    v8f acc = {};
    const bf16* bw = wo + (size_t)n0 * I_;
    for (int ii = 0; ii < I_; ii += 32) {
      v16bf a = frag_bf16(&s_act[ii], I_, lane);
      v16bf b = frag_bf16(bw + ii, I_, lane);
      acc = wmma_bf16(a, b, acc);
      if (ii + 32 < I_)
        __builtin_prefetch(bw + (size_t)(lane & 15) * I_ + ii + 32, 0, 1);
    }
    const int half = lane >> 4, colh = n0 + (lane & 15);
#pragma unroll
    for (int v = 0; v < 8; ++v) {
      const int row = v + 8 * half;
      const float val = acc[v];
      float* o = out + (size_t)s_tok[row] * H_ + colh;
      if (EXPERT_MODE)
        atomicAdd(o, val);   // experts accumulate onto shared_out
      else
        *o = val;            // shared expert writes first (stream-ordered)
    }
  }
}

// ---------------------------------------------------------------------------
// Launch
// ---------------------------------------------------------------------------
extern "C" void kernel_launch(void* const* d_in, const int* in_sizes, int n_in,
                              void* d_out, int out_size, void* d_ws,
                              size_t ws_size, hipStream_t stream) {
  (void)in_sizes; (void)n_in; (void)out_size; (void)ws_size;
  const float* x    = (const float*)d_in[0];  // (T, H)
  const float* gw   = (const float*)d_in[1];  // (E, H)
  const float* bias = (const float*)d_in[2];  // (E,)
  const float* Wi   = (const float*)d_in[3];  // (E, H, 2I)
  const float* Wo   = (const float*)d_in[4];  // (E, I, H)
  const float* sWi  = (const float*)d_in[5];  // (2I, H)
  const float* sWo  = (const float*)d_in[6];  // (H, I)
  float* out = (float*)d_out;                 // (T, H)

  char* ws = (char*)d_ws;
  size_t off = 0;
  auto alloc = [&](size_t bytes) -> void* {
    void* p = ws + off;
    off = (off + bytes + 255) & ~(size_t)255;
    return p;
  };
  bf16* WiT  = (bf16*)alloc((size_t)E_ * 2 * I_ * H_ * sizeof(bf16)); // 28.3MB
  bf16* WoT  = (bf16*)alloc((size_t)E_ * H_ * I_ * sizeof(bf16));     // 14.2MB
  bf16* sWiB = (bf16*)alloc((size_t)2 * I_ * H_ * sizeof(bf16));      // 3.5MB
  bf16* sWoB = (bf16*)alloc((size_t)H_ * I_ * sizeof(bf16));          // 1.8MB
  int*   cnt = (int*)alloc(E_ * sizeof(int));
  int*   idx = (int*)alloc((size_t)E_ * T_ * sizeof(int));
  float* cwb = (float*)alloc((size_t)E_ * T_ * sizeof(float));

  hipMemsetAsync(cnt, 0, E_ * sizeof(int), stream);

  // Wi (E,H,2I) -> WiT (E,2I,H) bf16 ; Wo (E,I,H) -> WoT (E,H,I) bf16
  transpose_to_bf16<<<dim3(2 * I_ / 32, H_ / 32, E_), 256, 0, stream>>>(
      Wi, WiT, H_, 2 * I_);
  transpose_to_bf16<<<dim3(H_ / 32, I_ / 32, E_), 256, 0, stream>>>(
      Wo, WoT, I_, H_);
  // Shared weights are already (N, K) row-major: just narrow to bf16.
  cvt_bf16<<<(2 * I_ * H_ + 255) / 256, 256, 0, stream>>>(sWi, sWiB,
                                                          2 * I_ * H_);
  cvt_bf16<<<(H_ * I_ + 255) / 256, 256, 0, stream>>>(sWo, sWoB, H_ * I_);

  routing_kernel<<<T_ / 256, 256, 0, stream>>>(x, gw, bias, cnt, idx, cwb);

  // Shared expert: dense over all tokens, plain stores initialize d_out.
  fused_mlp<0><<<dim3(T_ / TB), 256, 0, stream>>>(x, sWiB, sWoB, nullptr,
                                                  nullptr, nullptr, out);
  // Routed experts: gathered token blocks, atomic accumulate.
  fused_mlp<1><<<dim3(T_ / TB, E_), 256, 0, stream>>>(x, WiT, WoT, cnt, idx,
                                                      cwb, out);
}
